// Jumper_6700148982313
// MI455X (gfx1250) — matrix operand order repaired and checked by reading, not compiled, not run on hardware
//
#include <hip/hip_runtime.h>
#include <hip/hip_bf16.h>

// ---------------------------------------------------------------------------
// Jumper model for MI455X (gfx1250, wave32).
//  * GEMMs on v_wmma_f32_16x16x32_f16 (weights converted to f16 once per call).
//  * Weight slabs staged into LDS by the Tensor Data Mover (TENSOR_LOAD_TO_LDS,
//    TENSORcnt) — removes the 4x B-operand redundancy across M-tiles and keeps
//    the hot loads on LDS bandwidth; A keeps streaming from global/L2.
//  * Sequential scan (N=20) expressed as small kernels so graph replay
//    amortizes launch latency; never materializes the 268MB [B,S,2H] input.
// ---------------------------------------------------------------------------

#define Bsz   64
#define Ssz   2048
#define Hsz   512
#define H2    1024      // 2*H
#define H4    2048      // 4*H
#define Kp1   11        // K+1
#define Nstep 20

typedef __attribute__((ext_vector_type(16))) _Float16 v16h;
typedef __attribute__((ext_vector_type(8)))  float    v8f;
typedef unsigned int u32x4 __attribute__((ext_vector_type(4)));
typedef int          i32x4 __attribute__((ext_vector_type(4)));
typedef int          i32x8 __attribute__((ext_vector_type(8)));

__device__ __forceinline__ float sigm(float x) { return 1.f / (1.f + expf(-x)); }

__device__ __forceinline__ unsigned hash_u32(unsigned x) {
    x ^= x >> 16; x *= 0x7feb352dU;
    x ^= x >> 15; x *= 0x846ca68bU;
    x ^= x >> 16; return x;
}

// ---------------------------------------------------------------------------
// TDM: DMA a 16-row x rowElems (f16) tile from global into LDS at ldsOff.
// D# per CDNA5 ISA ch.8: group0 = {count/flags, lds_addr, global_addr, type=2},
// group1 = {data_size=2B, tensor_dim0/1, tile_dim0/1, dim0_stride}. 2-D tile,
// no gather/iterate/pad/multicast. Tracked by TENSORcnt.
// ---------------------------------------------------------------------------
__device__ __forceinline__ void tdm_load_16rows(unsigned ldsOff,
                                                const _Float16* __restrict__ gsrc,
                                                unsigned rowElems)
{
    unsigned long long ga = (unsigned long long)(uintptr_t)gsrc;
    u32x4 g0;
    g0[0] = 1u;                                            // count=1 (valid), user mode
    g0[1] = ldsOff;                                        // lds_addr (bytes)
    g0[2] = (unsigned)(ga & 0xffffffffu);                  // global_addr[31:0]
    g0[3] = (unsigned)((ga >> 32) & 0x01ffffffu) | (2u << 30); // addr[56:32] | type=2
    i32x8 g1;
    g1[0] = (int)(1u << 16);                               // data_size=1 -> 2 bytes
    g1[1] = (int)(rowElems << 16);                         // tensor_dim0[15:0] @bits63:48
    g1[2] = (int)((rowElems >> 16) | (16u << 16));         // dim0[31:16] | tensor_dim1=16
    g1[3] = (int)(rowElems << 16);                         // dim1[31:16]=0 | tile_dim0
    g1[4] = 16;                                            // tile_dim1=16, tile_dim2=0
    g1[5] = (int)rowElems;                                 // tensor_dim0_stride[31:0]
    g1[6] = 0;                                             // stride0[47:32] | dim1_stride lo
    g1[7] = 0;
    i32x4 gz = {0, 0, 0, 0};
#if __clang_major__ >= 23
    i32x8 gz8 = {0, 0, 0, 0, 0, 0, 0, 0};
    __builtin_amdgcn_tensor_load_to_lds(g0, g1, gz, gz, gz8, 0);
#else
    __builtin_amdgcn_tensor_load_to_lds(g0, g1, gz, gz, 0);
#endif
}

// ---------------------------------------------------------------------------
// One 16x16 f32 output tile of  C += A * W^T  accumulated over K:
//   A: f16 row-major [M x K], lda elems; rows start at m0 (global)
//   W: f16 row-major [16 x K] slab (LDS or global); B[k][n] = W[n][k]
// Per-lane layouts follow CDNA5 ISA 7.12.2 (16-bit A 16x32; B 32x16; f32 C/D).
// ---------------------------------------------------------------------------
__device__ __forceinline__ void wmma_tile_acc(const _Float16* __restrict__ A, int lda, int m0,
                                              const _Float16* __restrict__ W, int ldw,
                                              int K, v8f& acc, int lane)
{
    const int khalf = (lane >> 4);              // 0: lanes 0-15, 1: lanes 16-31
    const _Float16* aRow = A + (size_t)(m0 + (lane & 15)) * lda;
    const _Float16* wRow = W + (size_t)(lane & 15) * ldw;
    __builtin_prefetch(aRow, 0, 1);
    for (int k = 0; k < K; k += 32) {
        v16h a, b;
#pragma unroll
        for (int v = 0; v < 8; ++v) {           // A 16x32: VGPR v holds a K pair
            int k0 = k + ((v < 4) ? (2 * v) : (16 + 2 * (v - 4))) + 8 * khalf;
            a[2 * v]     = aRow[k0];
            a[2 * v + 1] = aRow[k0 + 1];
        }
#pragma unroll
        for (int v = 0; v < 8; ++v) {           // B 32x16: VGPR v holds a K pair (same order as A)
            int k0 = k + ((v < 4) ? (2 * v) : (16 + 2 * (v - 4))) + 8 * khalf;
            b[2 * v]     = wRow[k0];
            b[2 * v + 1] = wRow[k0 + 1];
        }
        acc = __builtin_amdgcn_wmma_f32_16x16x32_f16(
                  false, a, false, b, (short)0, acc, false, false);
    }
}

// --------------------------- weight conversion -----------------------------
__global__ void k_f32_to_f16(const float* __restrict__ src, _Float16* __restrict__ dst, int n)
{
    int i = blockIdx.x * blockDim.x + threadIdx.x;
    if (i < n) dst[i] = (_Float16)src[i];
}

// ------------------------------- init --------------------------------------
__global__ void k_init(const float* __restrict__ emb, const int* __restrict__ questions,
                       const int* __restrict__ masks,
                       float* Qf, _Float16* Q16, float* h, float* c, float* hiddens,
                       _Float16* h16, int* rows, int* hset, int* lengths, int* last_rows,
                       int* doneFlag, int* doneCount)
{
    __shared__ int red[256];
    const int b = blockIdx.x, t = threadIdx.x;
    int s = 0;
    for (int j = t; j < Ssz; j += 256) s += (masks[b * Ssz + j] == 0);
    red[t] = s; __syncthreads();
    for (int st = 128; st; st >>= 1) { if (t < st) red[t] += red[t + st]; __syncthreads(); }
    if (t == 0) {
        lengths[b] = red[0]; last_rows[b] = red[0] - 1;
        rows[b] = 0; hset[b] = 0;
        if (b == 0) { *doneFlag = 0; *doneCount = 0; }
    }
    const int tok = questions[b];
    for (int j = t; j < Hsz; j += 256) {
        float q = emb[(size_t)tok * Hsz + j];
        Qf[b * Hsz + j] = q; Q16[b * Hsz + j] = (_Float16)q;
        h[b * Hsz + j] = 0.f; c[b * Hsz + j] = 0.f; hiddens[b * Hsz + j] = 0.f;
        h16[b * Hsz + j] = (_Float16)0.f;
    }
}

// --------------------------- step: gather ----------------------------------
__global__ void k_step_gather(const float* __restrict__ emb, const int* __restrict__ contexts,
                              const int* __restrict__ masks, const _Float16* __restrict__ Q16,
                              const float* __restrict__ h, float* hiddens, int* hset,
                              const int* __restrict__ rows, const int* __restrict__ lengths,
                              const int* __restrict__ last_rows,
                              _Float16* x16, int* rows2_out, int* deadArr,
                              const int* __restrict__ doneFlag, int n)
{
    const int b = blockIdx.x, t = threadIdx.x;
    if (*doneFlag) { if (t == 0) deadArr[n * Bsz + b] = 1; return; }
    const int row = rows[b], len = lengths[b];
    const int fp = (row >= len);
    const int row2 = fp ? last_rows[b] : row;
    const int newly = fp && (hset[b] == 0);
    if (t == 0) { rows2_out[b] = row2; deadArr[n * Bsz + b] = fp; }
    if (newly)
        for (int j = t; j < Hsz; j += 256) hiddens[b * Hsz + j] = h[b * Hsz + j];
    __syncthreads();
    if (t == 0 && fp) hset[b] = 1;
    const int tok   = contexts[b * Ssz + row2];
    const int valid = (masks[b * Ssz + row2] == 0);
    const float* e = emb + (size_t)tok * Hsz;
    for (int j = t; j < Hsz; j += 256) {
        x16[b * H2 + j]       = valid ? (_Float16)e[j]   : (_Float16)0.f;
        x16[b * H2 + Hsz + j] = valid ? Q16[b * Hsz + j] : (_Float16)0.f;
    }
}

// --------------------------- step: LSTM GEMM (WMMA + TDM) ------------------
// g[64,2048] = x16[64,1024] @ W_ih^T + h16[64,512] @ W_hh^T
// 128 blocks (one 16-col slice each) x 4 wave32 (one 16-row tile each).
// Wave 0 TDM-stages the block's 16xK weight slab (48KB) into LDS once.
__global__ void __launch_bounds__(128)
k_gemm_lstm(const _Float16* __restrict__ x16, const _Float16* __restrict__ h16,
            const _Float16* __restrict__ Wih16, const _Float16* __restrict__ Whh16,
            float* __restrict__ g, const int* __restrict__ doneFlag)
{
    __shared__ __align__(16) _Float16 sW[16 * H2 + 16 * Hsz];   // 48 KB
    if (*doneFlag) return;                              // block-uniform, EXEC stays full
    const int nt   = blockIdx.x;                        // 0..127 column tile
    const int mt   = threadIdx.x >> 5;                  // 0..3   row tile (one per wave)
    const int lane = threadIdx.x & 31;
    if (threadIdx.x < 32) {                             // wave 0 drives the TDM
        const unsigned base = (unsigned)(uintptr_t)&sW[0];
        tdm_load_16rows(base,                Wih16 + (size_t)nt * 16 * H2,  H2);
        tdm_load_16rows(base + 16 * H2 * 2,  Whh16 + (size_t)nt * 16 * Hsz, Hsz);
        __builtin_amdgcn_s_wait_tensorcnt(0);
    }
    __syncthreads();                                    // publish LDS slab to all waves
    v8f acc = {};
    wmma_tile_acc(x16, H2,  mt * 16, sW,           H2,  H2,  acc, lane);
    wmma_tile_acc(h16, Hsz, mt * 16, sW + 16 * H2, Hsz, Hsz, acc, lane);
    const int mBase = mt * 16 + ((lane >> 4) ? 8 : 0);  // f32 C/D layout
    const int ncol  = nt * 16 + (lane & 15);
#pragma unroll
    for (int r = 0; r < 8; ++r)
        g[(size_t)(mBase + r) * H4 + ncol] = acc[r];
}

// --------------- step: gates, logits, sample, state update ------------------
__global__ void k_step_point(const float* __restrict__ g, const float* __restrict__ b_ih,
                             const float* __restrict__ b_hh,
                             float* h, float* c, _Float16* h16,
                             const float* __restrict__ W_lin, const float* __restrict__ b_lin,
                             const float* __restrict__ W_base, const float* __restrict__ b_base,
                             int* rows, const int* __restrict__ rows2,
                             const int* __restrict__ lengths,
                             float* lp, float* bl, int* valids,
                             const int* __restrict__ doneFlag, int* doneCount, int n)
{
    __shared__ float h2s[Hsz];
    __shared__ float dots[12];
    const int b = blockIdx.x, t = threadIdx.x;          // 256 threads
    if (*doneFlag) {
        if (t == 0) { lp[n * Bsz + b] = 0.f; bl[n * Bsz + b] = 0.f;
                      if (b == 0) valids[n] = 0; }
        return;
    }
    for (int j = t; j < Hsz; j += 256) {
        float gi = g[b * H4 + j]           + b_ih[j]           + b_hh[j];
        float gf = g[b * H4 + Hsz + j]     + b_ih[Hsz + j]     + b_hh[Hsz + j];
        float gg = g[b * H4 + 2 * Hsz + j] + b_ih[2 * Hsz + j] + b_hh[2 * Hsz + j];
        float go = g[b * H4 + 3 * Hsz + j] + b_ih[3 * Hsz + j] + b_hh[3 * Hsz + j];
        float c2 = sigm(gf) * c[b * Hsz + j] + sigm(gi) * tanhf(gg);
        float h2 = sigm(go) * tanhf(c2);
        c[b * Hsz + j] = c2; h[b * Hsz + j] = h2;
        h16[b * Hsz + j] = (_Float16)h2; h2s[j] = h2;
    }
    __syncthreads();
    const int wv = t >> 5, lane = t & 31;               // 8 wave32s; 12 dots
    for (int j = wv; j < 12; j += 8) {
        const float* wr = (j < Kp1) ? (W_lin + j * Hsz) : W_base;
        float s = 0.f;
        for (int kk = lane; kk < Hsz; kk += 32) s += h2s[kk] * wr[kk];
        for (int off = 16; off; off >>= 1) s += __shfl_down(s, off, 32);
        if (lane == 0) dots[j] = s;
    }
    __syncthreads();
    if (t == 0) {
        float logit[Kp1], mx = -1e30f;
        for (int j = 0; j < Kp1; ++j) { logit[j] = dots[j] + b_lin[j]; mx = fmaxf(mx, logit[j]); }
        float Z = 0.f;
        for (int j = 0; j < Kp1; ++j) Z += expf(logit[j] - mx);
        const float logZ = logf(Z) + mx;
        // deterministic per-(step,batch) uniform -> CDF categorical sample
        const float u = (hash_u32(0x9E3779B9u ^ (unsigned)(n * 9781 + b * 6271)) >> 8)
                        * (1.0f / 16777216.0f);
        int jump = Kp1 - 1; float cdf = 0.f;
        for (int j = 0; j < Kp1; ++j) { cdf += expf(logit[j] - logZ);
                                        if (u < cdf) { jump = j; break; } }
        const float logp = logit[jump] - logZ;
        const float base = dots[Kp1] + b_base[0];
        const int r4 = rows2[b] + 1 + jump;
        rows[b] = r4;
        lp[n * Bsz + b] = logp; bl[n * Bsz + b] = base;
        if (r4 >= lengths[b]) atomicAdd(doneCount, 1);
        if (b == 0) valids[n] = 1;
    }
}

__global__ void k_step_done(int* doneFlag, int* doneCount)
{
    if (!*doneFlag && *doneCount == Bsz) *doneFlag = 1;
    *doneCount = 0;
}

// --------------------------- final: build data ------------------------------
__global__ void k_build_data(const float* __restrict__ Qf, const float* __restrict__ h,
                             const float* __restrict__ hiddens, const int* __restrict__ hset,
                             _Float16* data16)
{
    const int b = blockIdx.x, t = threadIdx.x;
    const float* P = hset[b] ? (hiddens + b * Hsz) : (h + b * Hsz);
    for (int j = t; j < Hsz; j += 256) {
        float q = Qf[b * Hsz + j], p = P[j];
        data16[b * H4 + j]           = (_Float16)q;
        data16[b * H4 + Hsz + j]     = (_Float16)p;
        data16[b * H4 + 2 * Hsz + j] = (_Float16)(q - p);
        data16[b * H4 + 3 * Hsz + j] = (_Float16)(q * p);
    }
}

// ------------------- final: MLP GEMM (WMMA + TDM) + leaky ReLU --------------
// s1[64,256] = leaky_relu(data16[64,2048] @ W1^T + b1).
// 16 blocks (one 16-col slice, 64KB LDS slab via TDM) x 4 wave32.
__global__ void __launch_bounds__(128)
k_gemm_mlp(const _Float16* __restrict__ data16, const _Float16* __restrict__ W1_16,
           const float* __restrict__ b1, float* __restrict__ s1)
{
    __shared__ __align__(16) _Float16 sW[16 * H4];              // 64 KB
    const int nt   = blockIdx.x;                        // 0..15
    const int mt   = threadIdx.x >> 5;                  // 0..3
    const int lane = threadIdx.x & 31;
    if (threadIdx.x < 32) {
        tdm_load_16rows((unsigned)(uintptr_t)&sW[0], W1_16 + (size_t)nt * 16 * H4, H4);
        __builtin_amdgcn_s_wait_tensorcnt(0);
    }
    __syncthreads();
    v8f acc = {};
    wmma_tile_acc(data16, H4, mt * 16, sW, H4, H4, acc, lane);
    const int mBase = mt * 16 + ((lane >> 4) ? 8 : 0);
    const int ncol  = nt * 16 + (lane & 15);
#pragma unroll
    for (int r = 0; r < 8; ++r) {
        float v = acc[r] + b1[ncol];
        s1[(size_t)(mBase + r) * 256 + ncol] = (v > 0.f) ? v : 0.01f * v;
    }
}

// --------------------------- final: all losses ------------------------------
__global__ void k_final_loss(const float* __restrict__ s1, const float* __restrict__ W2,
                             const float* __restrict__ b2, const int* __restrict__ labels,
                             const float* __restrict__ lp, const float* __restrict__ bl,
                             const int* __restrict__ deadArr, const int* __restrict__ valids,
                             float* __restrict__ out)
{
    __shared__ float sc[Bsz];
    __shared__ float red[256];
    const int t = threadIdx.x;
    if (t < Bsz) {
        const float* r = s1 + t * 256;
        float s = 0.f;
        for (int k = 0; k < 256; ++k) s += r[k] * W2[k];
        sc[t] = sigm(s + b2[0]);
    }
    __syncthreads();
    int nv = 0;
#pragma unroll
    for (int n = 0; n < Nstep; ++n) nv += valids[n];
    const float reward_a = -(float)nv / (float)Ssz;

    float sre = 0.f, smse = 0.f;
    for (int i = t; i < Nstep * Bsz; i += 256) {
        if (deadArr[i] == 0) {
            const int b = i & (Bsz - 1);
            float rr = (((sc[b] > 0.5f) ? 1 : 0) == labels[b]) ? 1.f : 0.f;
            float rw = rr + reward_a;
            float l = lp[i], bb = bl[i];
            sre  += (rw - bb) * l;
            smse += (bb - rw) * (bb - rw);
        }
    }
    float sf = 0.f;
    if (t < Bsz) {
        float y = (float)labels[t], p = sc[t];
        float bce = -(y * logf(fmaxf(p, 1e-12f)) + (1.f - y) * logf(fmaxf(1.f - p, 1e-12f)));
        float pt = expf(-bce);
        sf = (1.f - pt) * (1.f - pt) * bce;
    }
    red[t] = sre; __syncthreads();
    for (int st = 128; st; st >>= 1) { if (t < st) red[t] += red[t + st]; __syncthreads(); }
    const float reinf = red[0]; __syncthreads();
    red[t] = smse; __syncthreads();
    for (int st = 128; st; st >>= 1) { if (t < st) red[t] += red[t + st]; __syncthreads(); }
    const float mse = red[0]; __syncthreads();
    red[t] = sf; __syncthreads();
    for (int st = 128; st; st >>= 1) { if (t < st) red[t] += red[t + st]; __syncthreads(); }
    if (t == 0) {
        const float denom = (float)nv * (float)Bsz;
        out[0] = red[0] - reinf / denom + mse / denom;
    }
}

// ---------------------------------------------------------------------------
extern "C" void kernel_launch(void* const* d_in, const int* in_sizes, int n_in,
                              void* d_out, int out_size, void* d_ws, size_t ws_size,
                              hipStream_t stream)
{
    (void)in_sizes; (void)n_in; (void)out_size; (void)ws_size;
    const float* emb    = (const float*)d_in[0];
    const float* W_ih   = (const float*)d_in[1];
    const float* W_hh   = (const float*)d_in[2];
    const float* b_ih   = (const float*)d_in[3];
    const float* b_hh   = (const float*)d_in[4];
    const float* W_lin  = (const float*)d_in[5];
    const float* b_lin  = (const float*)d_in[6];
    const float* W_base = (const float*)d_in[7];
    const float* b_base = (const float*)d_in[8];
    const float* W1     = (const float*)d_in[9];
    const float* b1     = (const float*)d_in[10];
    const float* W2     = (const float*)d_in[11];
    const float* b2     = (const float*)d_in[12];
    const int* questions = (const int*)d_in[13];
    const int* contexts  = (const int*)d_in[14];
    const int* masks     = (const int*)d_in[15];
    const int* labels    = (const int*)d_in[16];
    float* out = (float*)d_out;

    char* ws = (char*)d_ws;
    size_t off = 0;
    auto alloc = [&](size_t bytes) -> void* {
        off = (off + 255) & ~(size_t)255;
        void* p = ws + off; off += bytes; return p;
    };

    _Float16* Wih16  = (_Float16*)alloc((size_t)H4 * H2 * 2);
    _Float16* Whh16  = (_Float16*)alloc((size_t)H4 * Hsz * 2);
    _Float16* W1_16  = (_Float16*)alloc((size_t)256 * H4 * 2);
    _Float16* Q16    = (_Float16*)alloc((size_t)Bsz * Hsz * 2);
    _Float16* x16    = (_Float16*)alloc((size_t)Bsz * H2 * 2);
    _Float16* h16    = (_Float16*)alloc((size_t)Bsz * Hsz * 2);
    _Float16* data16 = (_Float16*)alloc((size_t)Bsz * H4 * 2);
    float* Qf      = (float*)alloc((size_t)Bsz * Hsz * 4);
    float* h       = (float*)alloc((size_t)Bsz * Hsz * 4);
    float* c       = (float*)alloc((size_t)Bsz * Hsz * 4);
    float* hiddens = (float*)alloc((size_t)Bsz * Hsz * 4);
    float* g       = (float*)alloc((size_t)Bsz * H4 * 4);
    float* s1      = (float*)alloc((size_t)Bsz * 256 * 4);
    float* lp      = (float*)alloc((size_t)Nstep * Bsz * 4);
    float* bl      = (float*)alloc((size_t)Nstep * Bsz * 4);
    int* deadArr   = (int*)alloc((size_t)Nstep * Bsz * 4);
    int* valids    = (int*)alloc((size_t)Nstep * 4);
    int* rows      = (int*)alloc(Bsz * 4);
    int* rows2     = (int*)alloc(Bsz * 4);
    int* lengths   = (int*)alloc(Bsz * 4);
    int* last_rows = (int*)alloc(Bsz * 4);
    int* hset      = (int*)alloc(Bsz * 4);
    int* doneFlag  = (int*)alloc(4);
    int* doneCount = (int*)alloc(4);

    // Weight conversion (f32 -> f16) each call; weights then live in L2.
    {
        int n0 = H4 * H2, n1 = H4 * Hsz, n2 = 256 * H4;
        k_f32_to_f16<<<(n0 + 255) / 256, 256, 0, stream>>>(W_ih, Wih16, n0);
        k_f32_to_f16<<<(n1 + 255) / 256, 256, 0, stream>>>(W_hh, Whh16, n1);
        k_f32_to_f16<<<(n2 + 255) / 256, 256, 0, stream>>>(W1,   W1_16, n2);
    }

    k_init<<<Bsz, 256, 0, stream>>>(emb, questions, masks, Qf, Q16, h, c, hiddens,
                                    h16, rows, hset, lengths, last_rows,
                                    doneFlag, doneCount);

    for (int n = 0; n < Nstep; ++n) {
        k_step_gather<<<Bsz, 256, 0, stream>>>(emb, contexts, masks, Q16, h, hiddens,
                                               hset, rows, lengths, last_rows,
                                               x16, rows2, deadArr, doneFlag, n);
        k_gemm_lstm<<<128, 128, 0, stream>>>(x16, h16, Wih16, Whh16, g, doneFlag);
        k_step_point<<<Bsz, 256, 0, stream>>>(g, b_ih, b_hh, h, c, h16,
                                              W_lin, b_lin, W_base, b_base,
                                              rows, rows2, lengths,
                                              lp, bl, valids, doneFlag, doneCount, n);
        k_step_done<<<1, 1, 0, stream>>>(doneFlag, doneCount);
    }

    k_build_data<<<Bsz, 256, 0, stream>>>(Qf, h, hiddens, hset, data16);
    k_gemm_mlp<<<16, 128, 0, stream>>>(data16, W1_16, b1, s1);
    k_final_loss<<<1, 256, 0, stream>>>(s1, W2, b2, labels, lp, bl, deadArr, valids, out);
}
